// Attention_17325898072716
// MI455X (gfx1250) — matrix operand order repaired and checked by reading, not compiled, not run on hardware
//
#include <hip/hip_runtime.h>
#include <hip/hip_bf16.h>

// ---------------------------------------------------------------------------
// GQA causal attention for MI455X (gfx1250, wave32, WMMA, async-LDS staging).
//   B=1, S=2048, H=1024, NH=16, NKV=4, HD=64, scale = 1/sqrt(H) = 1/32
// convert(fp32->f16) -> QKV GEMM (wmma, LDS-staged B) -> flash attn (wmma,
// LDS-staged K/V shared by the 4 heads of one KV group).
// ---------------------------------------------------------------------------

#define S_LEN   2048
#define HID     1024
#define NH      16
#define NKV     4
#define HD      64
#define O_TOT   (NH*HD + 2*NKV*HD)   // 1536 rows of fused [Wq;Wk;Wv]

typedef __attribute__((ext_vector_type(16))) _Float16 v16h;
typedef __attribute__((ext_vector_type(8)))  float    v8f;

// ---- CDNA5 async global->LDS path (guarded; fallback: sync staging) -------
#if defined(__AMDGCN__) && __has_builtin(__builtin_amdgcn_global_load_async_to_lds_b128) && __has_builtin(__builtin_amdgcn_s_wait_asynccnt)
#define ASYNC_LDS 1
#else
#define ASYNC_LDS 0
#endif

#if ASYNC_LDS
// Builtin signature (from hipcc diagnostic): params are pointers to a 16-byte
// GCC-vector of int, in AS(1) (printed "__device__") / AS(3) respectively.
typedef int b128_t __attribute__((vector_size(16)));
typedef __attribute__((address_space(1))) b128_t* g_b128_p;
typedef __attribute__((address_space(3))) b128_t* l_b128_p;
#endif

// Copy 16 bytes global->LDS at compile-time byte offset OFF (applies to both
// sides, matching the ISA's INST_OFFSET semantics for async loads).
template <int OFF>
__device__ __forceinline__ void lds_stage16(const _Float16* g, _Float16* l) {
#if ASYNC_LDS
  // generic->AS casts via integers: AS(3) pointers are the low 32 bits of a
  // generic LDS pointer (aperture scheme), AS(1) is the full 64-bit address.
  __builtin_amdgcn_global_load_async_to_lds_b128(
      (g_b128_p)(unsigned long long)(size_t)g,
      (l_b128_p)(unsigned int)(size_t)l,
      OFF, 0);
#else
  *reinterpret_cast<int4*>(reinterpret_cast<char*>(l) + OFF) =
      *reinterpret_cast<const int4*>(reinterpret_cast<const char*>(g) + OFF);
#endif
}

__device__ __forceinline__ void async_wait_barrier() {
#if ASYNC_LDS
  __builtin_amdgcn_s_wait_asynccnt(0);   // s_wait_asynccnt 0
#endif
  __syncthreads();
}

// ---- WMMA fragment loaders (contiguous k = 16*(lane/16)+s in A and B, wave32) -------------------

// A fragment: 16x32 f16, src row-major [16][ld] (M x K)
__device__ __forceinline__ v16h load_frag_a(const _Float16* __restrict__ src, int ld) {
  const int lane = threadIdx.x & 31;
  const int g = lane >> 4;
  const int m = lane & 15;
  const _Float16* row = src + m * ld;
  v16h a;
#pragma unroll
  for (int j = 0; j < 8; ++j) {
    const int k = 16 * g + 2 * j;
    a[2 * j]     = row[k];
    a[2 * j + 1] = row[k + 1];
  }
  return a;
}

// B fragment: 32x16 f16 (K x N), B[k][n] = src[n*ld + k]  (src is [N][K])
__device__ __forceinline__ v16h load_frag_b_nk(const _Float16* __restrict__ src, int ld) {
  const int lane = threadIdx.x & 31;
  const int g = lane >> 4;
  const int n = lane & 15;
  const _Float16* col = src + n * ld + 16 * g;
  v16h b;
#pragma unroll
  for (int j = 0; j < 8; ++j) {
    b[2 * j]     = col[2 * j];
    b[2 * j + 1] = col[2 * j + 1];
  }
  return b;
}

// B fragment: 32x16 f16 (K x N), B[k][n] = src[k*ld + n]  (src is [K][N])
__device__ __forceinline__ v16h load_frag_b_kn(const _Float16* __restrict__ src, int ld) {
  const int lane = threadIdx.x & 31;
  const int g = lane >> 4;
  const int n = lane & 15;
  v16h b;
#pragma unroll
  for (int j = 0; j < 8; ++j) {
    b[2 * j]     = src[(2 * j     + 16 * g) * ld + n];
    b[2 * j + 1] = src[(2 * j + 1 + 16 * g) * ld + n];
  }
  return b;
}

__device__ __forceinline__ v8f wmma_f16(v16h a, v16h b, v8f c) {
  return __builtin_amdgcn_wmma_f32_16x16x32_f16(false, a, false, b, (short)0, c,
                                                false, false);
}

// ---- Kernel 1: fp32 -> f16 conversion -------------------------------------

__global__ __launch_bounds__(256) void cvt_kernel(
    const float* __restrict__ x,  const float* __restrict__ Wq,
    const float* __restrict__ Wk, const float* __restrict__ Wv,
    _Float16* __restrict__ xh, _Float16* __restrict__ Wh) {
  const int NX  = S_LEN * HID;
  const int NWQ = NH  * HD * HID;
  const int NWK = NKV * HD * HID;
  const int NWV = NKV * HD * HID;
  const int total = NX + NWQ + NWK + NWV;
  for (int i = blockIdx.x * blockDim.x + threadIdx.x; i < total;
       i += gridDim.x * blockDim.x) {
    if (i < NX) {
      xh[i] = (_Float16)x[i];
    } else {
      const int j = i - NX;
      float v = (j < NWQ) ? Wq[j]
              : (j < NWQ + NWK) ? Wk[j - NWQ]
              : Wv[j - NWQ - NWK];
      Wh[j] = (_Float16)v;
    }
  }
}

// ---- Kernel 2: fused QKV projection GEMM ----------------------------------
// Block = 8 waves computing a 128(M) x 64(N) tile; the shared 64-wide B strip
// is staged chunk-by-chunk (64x32 f16 = 4KB) into double-buffered LDS with
// async loads. out[s,o] = sum_k xh[s,k]*Wh[o,k]; Q pre-scaled by 1/32.

__global__ __launch_bounds__(256) void qkv_gemm_kernel(
    const _Float16* __restrict__ xh, const _Float16* __restrict__ Wh,
    _Float16* __restrict__ Qh, _Float16* __restrict__ Kh,
    _Float16* __restrict__ Vh) {
  __shared__ _Float16 Bsm[2][64 * 32];   // 2 x 4KB

  const int tid  = threadIdx.x;
  const int wave = tid >> 5;
  const int lane = tid & 31;
  const int g = lane >> 4;
  const int n = lane & 15;

  const int NSTRIP = O_TOT / 64;         // 24
  const int mb = blockIdx.x / NSTRIP;    // 0..15  (128-row block)
  const int ns = blockIdx.x % NSTRIP;    // 0..23  (64-col strip)
  const int row0 = mb * 128 + wave * 16;

  // staging map: thread t -> output row so (0..63), 8-half segment sk
  const int so = tid >> 2;
  const int sk = (tid & 3) * 8;

  const _Float16* Ap = xh + (size_t)row0 * HID;
  v8f acc[4] = {};

  // prologue: stage k-chunk 0 into buffer 0
  lds_stage16<0>(Wh + (size_t)(ns * 64 + so) * HID + sk, &Bsm[0][so * 32 + sk]);
  async_wait_barrier();

  for (int it = 0; it < HID / 32; ++it) {
    const int k0 = it * 32;
    const int cur = it & 1;
    if (it + 1 < HID / 32) {   // stage next chunk while computing (uniform)
      lds_stage16<0>(Wh + (size_t)(ns * 64 + so) * HID + (k0 + 32) + sk,
                     &Bsm[cur ^ 1][so * 32 + sk]);
    }
    const v16h a = load_frag_a(Ap + k0, HID);
#pragma unroll
    for (int t = 0; t < 4; ++t) {
      acc[t] = wmma_f16(a, load_frag_b_nk(&Bsm[cur][(t * 16) * 32], 32), acc[t]);
    }
    async_wait_barrier();
  }

#pragma unroll
  for (int t = 0; t < 4; ++t) {
#pragma unroll
    for (int r = 0; r < 8; ++r) {
      const int srow = row0 + r + 8 * g;
      const int o = ns * 64 + t * 16 + n;
      const float c = acc[t][r];
      if (o < NH * HD) {
        Qh[((size_t)(o >> 6) * S_LEN + srow) * HD + (o & 63)] =
            (_Float16)(c * 0.03125f);             // fold softmax scale 1/32
      } else if (o < NH * HD + NKV * HD) {
        const int oo = o - NH * HD;
        Kh[((size_t)(oo >> 6) * S_LEN + srow) * HD + (oo & 63)] = (_Float16)c;
      } else {
        const int oo = o - NH * HD - NKV * HD;
        Vh[((size_t)(oo >> 6) * S_LEN + srow) * HD + (oo & 63)] = (_Float16)c;
      }
    }
  }
}

// ---- Kernel 3: causal flash attention -------------------------------------
// Block = 4 waves = the 4 Q-heads of one KV group at one 16-row query tile:
// identical causal trip counts (uniform barriers) and shared K/V chunks,
// staged 32 keys at a time (2 x 4KB) into double-buffered LDS (async).

__global__ __launch_bounds__(128) void attn_kernel(
    const _Float16* __restrict__ Qh, const _Float16* __restrict__ Kh,
    const _Float16* __restrict__ Vh, float* __restrict__ out) {
  __shared__ _Float16 Ksm[2][32 * 64];   // 2 x 4KB
  __shared__ _Float16 Vsm[2][32 * 64];   // 2 x 4KB
  __shared__ _Float16 Psm[4][16 * 32];   // 4 x 1KB (wave-private P staging)

  const int tid  = threadIdx.x;
  const int wave = tid >> 5;
  const int lane = tid & 31;
  const int g = lane >> 4;
  const int n = lane & 15;

  const int kvh = blockIdx.x >> 7;       // 0..3
  const int qt  = blockIdx.x & 127;      // query tile (same for all 4 waves)
  const int h   = kvh * 4 + wave;        // GQA: rep = NH/NKV = 4

  const _Float16* Qp  = Qh + ((size_t)h * S_LEN + qt * 16) * HD;
  const _Float16* Kp0 = Kh + (size_t)kvh * S_LEN * HD;
  const _Float16* Vp0 = Vh + (size_t)kvh * S_LEN * HD;

  const v16h aQ0 = load_frag_a(Qp, HD);        // d = 0..31
  const v16h aQ1 = load_frag_a(Qp + 32, HD);   // d = 32..63

  v8f acc[4] = {};
  float mrun[8], lrun[8];
#pragma unroll
  for (int r = 0; r < 8; ++r) { mrun[r] = -1e30f; lrun[r] = 0.0f; }

  _Float16* P = Psm[wave];

  // staging map: thread t -> key skey (0..31), 16-half segment at d = sd
  const int skey = tid >> 2;
  const int sd   = (tid & 3) * 16;

  const int nch = (16 * (qt + 1) + 31) >> 5;   // key chunks of 32 (uniform)

  // prologue: stage chunk 0
  {
    const _Float16* gK = Kp0 + (size_t)skey * HD + sd;
    const _Float16* gV = Vp0 + (size_t)skey * HD + sd;
    _Float16* lK = &Ksm[0][skey * 64 + sd];
    _Float16* lV = &Vsm[0][skey * 64 + sd];
    lds_stage16<0>(gK, lK);  lds_stage16<16>(gK, lK);
    lds_stage16<0>(gV, lV);  lds_stage16<16>(gV, lV);
  }
  async_wait_barrier();

  for (int kc = 0; kc < nch; ++kc) {
    const int key0 = kc * 32;
    const int cur = kc & 1;

    if (kc + 1 < nch) {   // stage next chunk while computing (uniform)
      const int nk0 = key0 + 32;
      const _Float16* gK = Kp0 + (size_t)(nk0 + skey) * HD + sd;
      const _Float16* gV = Vp0 + (size_t)(nk0 + skey) * HD + sd;
      _Float16* lK = &Ksm[cur ^ 1][skey * 64 + sd];
      _Float16* lV = &Vsm[cur ^ 1][skey * 64 + sd];
      lds_stage16<0>(gK, lK);  lds_stage16<16>(gK, lK);
      lds_stage16<0>(gV, lV);  lds_stage16<16>(gV, lV);
    }

    // ---- scores: S[16x32] = Q[16x64] @ K_chunk[64x32]^T (2 C-tiles) ----
    v8f s0 = {};
    v8f s1 = {};
    s0 = wmma_f16(aQ0, load_frag_b_nk(&Ksm[cur][0], 64), s0);
    s0 = wmma_f16(aQ1, load_frag_b_nk(&Ksm[cur][0] + 32, 64), s0);
    s1 = wmma_f16(aQ0, load_frag_b_nk(&Ksm[cur][16 * 64], 64), s1);
    s1 = wmma_f16(aQ1, load_frag_b_nk(&Ksm[cur][16 * 64] + 32, 64), s1);

    // ---- online softmax (row m lives across the 16 lanes of a half) ----
#pragma unroll
    for (int r = 0; r < 8; ++r) {
      const int m_row = qt * 16 + r + 8 * g;
      float v0 = s0[r];
      float v1 = s1[r];
      if (key0 + n > m_row)      v0 = -1e30f;   // causal mask (select only)
      if (key0 + 16 + n > m_row) v1 = -1e30f;

      float rm = fmaxf(v0, v1);
      rm = fmaxf(rm, __shfl_xor(rm, 1, 16));
      rm = fmaxf(rm, __shfl_xor(rm, 2, 16));
      rm = fmaxf(rm, __shfl_xor(rm, 4, 16));
      rm = fmaxf(rm, __shfl_xor(rm, 8, 16));

      const float mnew = fmaxf(mrun[r], rm);
      const float sc   = __expf(mrun[r] - mnew);
      const float p0   = __expf(v0 - mnew);
      const float p1   = __expf(v1 - mnew);

      float rs = p0 + p1;
      rs += __shfl_xor(rs, 1, 16);
      rs += __shfl_xor(rs, 2, 16);
      rs += __shfl_xor(rs, 4, 16);
      rs += __shfl_xor(rs, 8, 16);

      lrun[r] = lrun[r] * sc + rs;
      mrun[r] = mnew;

#pragma unroll
      for (int t = 0; t < 4; ++t) acc[t][r] *= sc;

      const int m = r + 8 * g;
      P[m * 32 + n]      = (_Float16)p0;
      P[m * 32 + 16 + n] = (_Float16)p1;
    }

    // LDS is in-order per wave: P stores complete before these loads.
    const v16h aP = load_frag_a(P, 32);

    // ---- O += P[16x32] @ V_chunk[32x64] ----
#pragma unroll
    for (int t = 0; t < 4; ++t) {
      acc[t] = wmma_f16(aP, load_frag_b_kn(&Vsm[cur][t * 16], 64), acc[t]);
    }

    async_wait_barrier();
  }

  // ---- normalize and write fp32 output [S][NH*HD] ----
#pragma unroll
  for (int t = 0; t < 4; ++t) {
#pragma unroll
    for (int r = 0; r < 8; ++r) {
      const int srow = qt * 16 + r + 8 * g;
      const int d = t * 16 + n;
      out[(size_t)srow * (NH * HD) + h * HD + d] = acc[t][r] / lrun[r];
    }
  }
}

// ---- launch ---------------------------------------------------------------

extern "C" void kernel_launch(void* const* d_in, const int* in_sizes, int n_in,
                              void* d_out, int out_size, void* d_ws, size_t ws_size,
                              hipStream_t stream) {
  const float* x  = (const float*)d_in[0];
  const float* Wq = (const float*)d_in[1];
  const float* Wk = (const float*)d_in[2];
  const float* Wv = (const float*)d_in[3];
  float* out = (float*)d_out;

  char* ws = (char*)d_ws;
  _Float16* xh = (_Float16*)(ws);                       // 4,194,304 B
  _Float16* Wh = (_Float16*)(ws + 4194304);             // 3,145,728 B
  _Float16* Qh = (_Float16*)(ws + 7340032);             // 4,194,304 B
  _Float16* Kh = (_Float16*)(ws + 11534336);            // 1,048,576 B
  _Float16* Vh = (_Float16*)(ws + 12582912);            // 1,048,576 B  (13.6 MB total)

  cvt_kernel<<<1024, 256, 0, stream>>>(x, Wq, Wk, Wv, xh, Wh);

  // 16 M-blocks x 24 N-strips, 8 waves/block
  qkv_gemm_kernel<<<384, 256, 0, stream>>>(xh, Wh, Qh, Kh, Vh);

  // 4 KV groups x 128 q-tiles, 4 waves/block (4 heads of the KV group)
  attn_kernel<<<512, 128, 0, stream>>>(Qh, Kh, Vh, out);
}
